// Encoder_25357486916224
// MI455X (gfx1250) — compile-verified, matrix operand order checked by reading, not run on hardware
//
#include <hip/hip_runtime.h>
#include <hip/hip_bf16.h>
#include <stdint.h>

#define IN_DIM_C   3000
#define NEG_SLOPE  0.2f
#define ETA_C      1e-6f

typedef __bf16  bf16_t;
typedef bf16_t  v16bf __attribute__((ext_vector_type(16)));
typedef float   v8f   __attribute__((ext_vector_type(8)));

union FragU { v16bf v; unsigned int u[8]; };

__device__ __forceinline__ void split2bf16(float x, unsigned short &hi, unsigned short &lo) {
  bf16_t h = (bf16_t)x;              // RNE to bf16 (top of mantissa)
  float hf = (float)h;
  bf16_t l = (bf16_t)(x - hf);       // residual
  hi = __builtin_bit_cast(unsigned short, h);
  lo = __builtin_bit_cast(unsigned short, l);
}

// Monotone float <-> uint mapping so unsigned atomicMax == float max
__device__ __forceinline__ unsigned int encodeF(float f) {
  unsigned int u = __builtin_bit_cast(unsigned int, f);
  return (u & 0x80000000u) ? ~u : (u | 0x80000000u);
}
__device__ __forceinline__ float decodeF(unsigned int u) {
  u = (u & 0x80000000u) ? (u & 0x7FFFFFFFu) : ~u;
  return __builtin_bit_cast(float, u);
}

__device__ __forceinline__ float softplusf(float x) {
  if (x > 20.f) return x;
  if (x < -20.f) return expf(x);
  return log1pf(expf(x));
}

// ---------------------------------------------------------------------------
// GEMM: C[M x Nc] = A[M x K] @ B[K x Nc], fp32 in/out, bf16x2 split WMMA core.
// Block: 256 threads (8 wave32). Block tile 128x64, K-step 32.
// Wave w computes rows [w*16, w*16+16) over 4 column tiles of 16.
// ---------------------------------------------------------------------------
#define BM 128
#define BN 64
#define KT 32
#define LDA 34   // 32 + 2 pad (17-bank stride, keeps 4B alignment of k-pairs)
#define LDB 34

__global__ void __launch_bounds__(256)
gemm_wmma_bf16x2(const float* __restrict__ A, const float* __restrict__ B,
                 float* __restrict__ C, int M, int K, int Nc)
{
  __shared__ unsigned short aHi[BM * LDA];
  __shared__ unsigned short aLo[BM * LDA];
  __shared__ unsigned short bHi[BN * LDB];
  __shared__ unsigned short bLo[BN * LDB];

  const int t    = threadIdx.x;
  const int wave = t >> 5;
  const int lane = t & 31;
  const int m0   = blockIdx.y * BM;
  const int n0   = blockIdx.x * BN;

  v8f acc[4];
  for (int i = 0; i < 4; ++i)
    for (int j = 0; j < 8; ++j) acc[i][j] = 0.f;

  for (int k0 = 0; k0 < K; k0 += KT) {
    // ---- stage A tile (128 x 32) as split bf16 ----
    for (int i = t; i < BM * KT; i += 256) {
      int r = i >> 5, c = i & 31;
      int gm = m0 + r, gk = k0 + c;
      float v = (gm < M && gk < K) ? A[(size_t)gm * K + gk] : 0.f;
      unsigned short h, l; split2bf16(v, h, l);
      aHi[r * LDA + c] = h;  aLo[r * LDA + c] = l;
    }
    // ---- stage B tile (32 x 64) transposed: lds[n][k] ----
    for (int i = t; i < KT * BN; i += 256) {
      int kr = i >> 6, nc = i & 63;
      int gk = k0 + kr;
      float v = (gk < K) ? B[(size_t)gk * Nc + n0 + nc] : 0.f;
      unsigned short h, l; split2bf16(v, h, l);
      bHi[nc * LDB + kr] = h;  bLo[nc * LDB + kr] = l;
    }
    __syncthreads();

    // speculative prefetch of next A tile (global_prefetch_b8)
    if (k0 + KT < K) {
      int pr = t >> 1;                  // 128 rows, 2 threads/row
      int gm = m0 + pr;
      if (gm < M) {
        int gk = k0 + KT + ((t & 1) << 4);
        if (gk < K) __builtin_prefetch(&A[(size_t)gm * K + gk], 0, 1);
      }
    }

    // ---- build A fragments (ISA 16-bit A 16x32 layout) ----
    const int row = wave * 16 + (lane & 15);
    const int kh  = (lane >> 4) * 8;       // k-half per lane group
    FragU ah, al;
    #pragma unroll
    for (int j = 0; j < 4; ++j) {
      ah.u[j]     = *(const unsigned int*)&aHi[row * LDA + kh + 2 * j];
      ah.u[j + 4] = *(const unsigned int*)&aHi[row * LDA + 16 + kh + 2 * j];
      al.u[j]     = *(const unsigned int*)&aLo[row * LDA + kh + 2 * j];
      al.u[j + 4] = *(const unsigned int*)&aLo[row * LDA + 16 + kh + 2 * j];
    }
    #pragma unroll
    for (int nt = 0; nt < 4; ++nt) {
      const int rb = nt * 16 + (lane & 15);
      FragU bh, bl;
      #pragma unroll
      for (int j = 0; j < 4; ++j) {
        bh.u[j]     = *(const unsigned int*)&bHi[rb * LDB + kh + 2 * j];
        bh.u[j + 4] = *(const unsigned int*)&bHi[rb * LDB + 16 + kh + 2 * j];
        bl.u[j]     = *(const unsigned int*)&bLo[rb * LDB + kh + 2 * j];
        bl.u[j + 4] = *(const unsigned int*)&bLo[rb * LDB + 16 + kh + 2 * j];
      }
      // hi*hi + hi*lo + lo*hi  (drop lo*lo ~ 2^-18 relative)
      acc[nt] = __builtin_amdgcn_wmma_f32_16x16x32_bf16(false, ah.v, false, bh.v,
                                                        (short)0, acc[nt], false, false);
      acc[nt] = __builtin_amdgcn_wmma_f32_16x16x32_bf16(false, ah.v, false, bl.v,
                                                        (short)0, acc[nt], false, false);
      acc[nt] = __builtin_amdgcn_wmma_f32_16x16x32_bf16(false, al.v, false, bh.v,
                                                        (short)0, acc[nt], false, false);
    }
    __syncthreads();
  }

  // ---- store: VGPR r -> row r (lanes 0-15) / row 8+r (lanes 16-31) ----
  const int mbase = m0 + wave * 16 + ((lane >> 4) << 3);
  const int ncol  = lane & 15;
  for (int nt = 0; nt < 4; ++nt) {
    #pragma unroll
    for (int r = 0; r < 8; ++r) {
      int gm = mbase + r;
      if (gm < M) C[(size_t)gm * Nc + n0 + nt * 16 + ncol] = acc[nt][r];
    }
  }
}

// ---------------------------------------------------------------------------
// Edge kernels (one wave per edge).  Edge list has E real edges + N self-loops.
// ---------------------------------------------------------------------------
__global__ void __launch_bounds__(256)
edge_scores_k(const float* __restrict__ xl, const float* __restrict__ xr,
              const int* __restrict__ ei, const float* __restrict__ att,
              float* __restrict__ eBuf, unsigned int* __restrict__ emaxEnc,
              int E, int nE, int H, int C)
{
  const int wave = threadIdx.x >> 5, lane = threadIdx.x & 31;
  const int e = blockIdx.x * 8 + wave;
  if (e >= nE) return;
  const int src = (e < E) ? ei[e]     : (e - E);
  const int dst = (e < E) ? ei[E + e] : (e - E);
  const int HC = H * C;
  for (int h = 0; h < H; ++h) {
    float part = 0.f;
    for (int c = lane; c < C; c += 32) {
      float v = xl[(size_t)src * HC + h * C + c] + xr[(size_t)dst * HC + h * C + c];
      float lr = (v > 0.f) ? v : NEG_SLOPE * v;
      part += att[h * C + c] * lr;
    }
    #pragma unroll
    for (int off = 16; off > 0; off >>= 1) part += __shfl_xor(part, off, 32);
    if (lane == 0) {
      eBuf[(size_t)e * H + h] = part;
      atomicMax(&emaxEnc[(size_t)dst * H + h], encodeF(part));
    }
  }
}

__global__ void __launch_bounds__(256)
edge_softmax_k(float* __restrict__ eBuf, const unsigned int* __restrict__ emaxEnc,
               const int* __restrict__ ei, float* __restrict__ denom,
               int E, int nE, int H)
{
  const int i = blockIdx.x * blockDim.x + threadIdx.x;
  if (i >= nE * H) return;
  const int e = i / H, h = i - e * H;
  const int dst = (e < E) ? ei[E + e] : (e - E);
  const float mx = decodeF(emaxEnc[(size_t)dst * H + h]);
  const float ee = expf(eBuf[i] - mx);
  eBuf[i] = ee;
  atomicAdd(&denom[(size_t)dst * H + h], ee);
}

__global__ void __launch_bounds__(256)
edge_aggregate_k(const float* __restrict__ xl, const float* __restrict__ eBuf,
                 const float* __restrict__ denom, const int* __restrict__ ei,
                 float* __restrict__ agg, int E, int nE, int H, int C)
{
  const int wave = threadIdx.x >> 5, lane = threadIdx.x & 31;
  const int e = blockIdx.x * 8 + wave;
  if (e >= nE) return;
  const int src = (e < E) ? ei[e]     : (e - E);
  const int dst = (e < E) ? ei[E + e] : (e - E);
  const int HC = H * C;
  for (int h = 0; h < H; ++h) {
    const float alpha = eBuf[(size_t)e * H + h] / denom[(size_t)dst * H + h];
    for (int c = lane; c < C; c += 32)
      atomicAdd(&agg[(size_t)dst * HC + h * C + c],
                alpha * xl[(size_t)src * HC + h * C + c]);
  }
}

__global__ void __launch_bounds__(256)
bias_act_k(float* __restrict__ y, const float* __restrict__ agg,
           const float* __restrict__ b, int Nn, int D, int relu)
{
  const int i = blockIdx.x * blockDim.x + threadIdx.x;
  if (i >= Nn * D) return;
  float v = agg[i] + b[i % D];
  if (relu && v < 0.f) v = 0.f;
  y[i] = v;
}

__global__ void __launch_bounds__(256)
fill_u32_k(unsigned int* __restrict__ p, size_t n, unsigned int v)
{
  size_t i = (size_t)blockIdx.x * blockDim.x + threadIdx.x;
  if (i < n) p[i] = v;
}

__global__ void __launch_bounds__(256)
head_out_k(const float* __restrict__ s, const float* __restrict__ p,
           float* __restrict__ out, int Nn)
{
  const int i = blockIdx.x * blockDim.x + threadIdx.x;
  if (i >= Nn * 64) return;
  const int n = i >> 6, c = i & 63;
  float mu, sv;
  if (c < 32) { mu = s[(size_t)n * 64 + c];        sv = s[(size_t)n * 64 + 32 + c]; }
  else        { mu = p[(size_t)n * 64 + (c - 32)]; sv = p[(size_t)n * 64 + (c - 32) + 32]; }
  out[i] = mu;
  out[(size_t)Nn * 64 + i] = softplusf(sv) + ETA_C;
}

// ---------------------------------------------------------------------------
extern "C" void kernel_launch(void* const* d_in, const int* in_sizes, int n_in,
                              void* d_out, int out_size, void* d_ws, size_t ws_size,
                              hipStream_t stream)
{
  const float* x  = (const float*)d_in[0];
  const int*   ei = (const int*)  d_in[1];
  const float* W[16];
  for (int i = 0; i < 16; ++i) W[i] = (const float*)d_in[2 + i];
  // W[0..3]=s1{Wl,Wr,att,b}  W[4..7]=s2  W[8..11]=p1  W[12..15]=p2

  const int Nn = in_sizes[0] / IN_DIM_C;   // 20000
  const int E  = in_sizes[1] / 2;          // 320000
  const int nE = E + Nn;                   // with self-loops

  float* ws = (float*)d_ws;
  size_t off = 0;
  float*        bufA   = ws + off; off += (size_t)Nn * 512;   // xl
  float*        bufB   = ws + off; off += (size_t)Nn * 512;   // xr
  float*        bufAgg = ws + off; off += (size_t)Nn * 512;   // agg / layer out
  float*        eBuf   = ws + off; off += (size_t)nE * 4;
  float*        denom  = ws + off; off += (size_t)Nn * 4;
  unsigned int* emax   = (unsigned int*)(ws + off); off += (size_t)Nn * 4;
  float*        sOut   = ws + off; off += (size_t)Nn * 64;
  float*        pOut   = ws + off; off += (size_t)Nn * 64;
  (void)ws_size; (void)n_in; (void)out_size;

  auto fill = [&](unsigned int* p, size_t n, unsigned int v) {
    fill_u32_k<<<(unsigned)((n + 255) / 256), 256, 0, stream>>>(p, n, v);
  };

  auto run_gat = [&](const float* xin, int K,
                     const float* Wl, const float* Wr, const float* att,
                     const float* bias, int H, int C, int relu, float* outBuf) {
    const int D = H * C;
    dim3 ggrid(D / 64, (Nn + BM - 1) / BM);
    gemm_wmma_bf16x2<<<ggrid, 256, 0, stream>>>(xin, Wl, bufA, Nn, K, D);
    gemm_wmma_bf16x2<<<ggrid, 256, 0, stream>>>(xin, Wr, bufB, Nn, K, D);
    fill(emax, (size_t)Nn * H, 0u);                 // encoded -inf floor
    fill((unsigned int*)denom, (size_t)Nn * H, 0u); // 0.0f
    fill((unsigned int*)bufAgg, (size_t)Nn * D, 0u);
    edge_scores_k<<<(nE + 7) / 8, 256, 0, stream>>>(bufA, bufB, ei, att, eBuf, emax,
                                                    E, nE, H, C);
    edge_softmax_k<<<(nE * H + 255) / 256, 256, 0, stream>>>(eBuf, emax, ei, denom,
                                                             E, nE, H);
    edge_aggregate_k<<<(nE + 7) / 8, 256, 0, stream>>>(bufA, eBuf, denom, ei, bufAgg,
                                                       E, nE, H, C);
    bias_act_k<<<(Nn * D + 255) / 256, 256, 0, stream>>>(outBuf, bufAgg, bias,
                                                         Nn, D, relu);
  };

  // ---- s path ----
  run_gat(x,      IN_DIM_C, W[0], W[1], W[2], W[3], 4, 128, 1, bufAgg); // s1 + relu
  run_gat(bufAgg, 512,      W[4], W[5], W[6], W[7], 1, 64,  0, sOut);   // s2
  // ---- p path ----
  run_gat(x,      IN_DIM_C, W[8],  W[9],  W[10], W[11], 4, 128, 0, bufAgg); // p1
  run_gat(bufAgg, 512,      W[12], W[13], W[14], W[15], 1, 64,  0, pOut);   // p2

  head_out_k<<<(Nn * 64 + 255) / 256, 256, 0, stream>>>(sOut, pOut, (float*)d_out, Nn);
}
